// SelfAttentionPytorch_51642686768040
// MI455X (gfx1250) — matrix operand order repaired and checked by reading, not compiled, not run on hardware
//
#include <hip/hip_runtime.h>
#include <hip/hip_bf16.h>

// ---------------------------------------------------------------------------
// tokens [1, L=2048, D=4096] f32, W_in [2048,2048] f32, W_out [2048,4096] f32
// Output: [2048] f32.
// All GEMMs in f16 via v_wmma_f32_16x16x32_f16 (f32 accumulate), LDS-tiled:
//   block tile 64x256 (8 waves x 32x64), double-buffered 20KB/k-step staging.
// ---------------------------------------------------------------------------

typedef _Float16 f16;
typedef __attribute__((ext_vector_type(4)))  _Float16 v4h;
typedef __attribute__((ext_vector_type(8)))  _Float16 v8h;
typedef __attribute__((ext_vector_type(16))) _Float16 v16h;
typedef __attribute__((ext_vector_type(8)))  float    v8f;

static constexpr int Lh = 2048;   // hidden length
static constexpr int Dc = 4096;   // channels

// ---------------- WMMA fragment loaders (CDNA5 wave32 layouts) -------------
// A 16x32 f16, row m = lane&15.
//   lanes 0-15 : K {0..7} then {16..23};  lanes 16-31: K {8..15} then {24..31}
__device__ __forceinline__ v16h load_frag_a(const f16* base, int lda,
                                            int row0, int k0, int lane) {
  const int m   = lane & 15;
  const int off = (lane < 16) ? 0 : 8;
  const f16* p = base + (size_t)(row0 + m) * lda + (k0 + off);
  v8h lo = *(const v8h*)(p);
  v8h hi = *(const v8h*)(p + 16);
  return __builtin_shufflevector(lo, hi, 0,1,2,3,4,5,6,7,8,9,10,11,12,13,14,15);
}

// B 32x16 f16 where B[k][n] = src[n0+n][k0+k] (src row-major over k).
//   lanes 0-15 : K {0..7} then {16..23};  lanes 16-31: K {8..15} then {24..31}; col n = lane&15.
__device__ __forceinline__ v16h load_frag_bt(const f16* base, int ldb,
                                             int n0, int k0, int lane) {
  const int n    = lane & 15;
  const int koff = (lane < 16) ? 0 : 8;
  const f16* p = base + (size_t)(n0 + n) * ldb + (k0 + koff);
  v8h lo = *(const v8h*)(p);
  v8h hi = *(const v8h*)(p + 16);
  return __builtin_shufflevector(lo, hi, 0,1,2,3,4,5,6,7,8,9,10,11,12,13,14,15);
}

// C/D 16x16 f32: VGPR r -> rows r / r+8 (lane<16 / >=16), col = lane&15.
__device__ __forceinline__ void store_tile_f16(f16* dst, int ldc,
                                               int row0, int col0, v8f c, int lane) {
  const int n  = lane & 15;
  const int rb = (lane < 16) ? 0 : 8;
#pragma unroll
  for (int r = 0; r < 8; ++r)
    dst[(size_t)(row0 + rb + r) * ldc + (col0 + n)] = (f16)c[r];
}

#define WMMA_F16(ACC, Af, Bf) \
  __builtin_amdgcn_wmma_f32_16x16x32_f16(false, (Af), false, (Bf), (short)0, (ACC), false, false)

// ---------------- LDS-tiled GEMM: C[M,N] = A[M,K] * B[N,K]^T ---------------
// grid: (N/256, M/64); block 256 threads (8 waves).
__global__ __launch_bounds__(256) void gemm_nt_f16_lds(const f16* __restrict__ A, int lda,
                                                       const f16* __restrict__ B, int ldb,
                                                       f16* __restrict__ C, int ldc, int K) {
  __shared__ __align__(16) f16 As[2][64 * 32];
  __shared__ __align__(16) f16 Bs[2][256 * 32];
  const int tid  = threadIdx.x;
  const int lane = tid & 31;
  const int w    = tid >> 5;
  const int rowB = blockIdx.y * 64;
  const int colB = blockIdx.x * 256;
  const int wr   = (w & 1) * 32;   // wave rows within block tile
  const int wc   = (w >> 1) * 64;  // wave cols within block tile

  // staging: A 64x32 -> thread t loads row t>>2, 8 cols; Bt 256x32 -> row t, 32 cols
  const int a_row = tid >> 2, a_col = (tid & 3) * 8;
  const f16* gA = A + (size_t)(rowB + a_row) * lda + a_col;
  const f16* gB = B + (size_t)(colB + tid) * ldb;

  v8h ra, rb0, rb1, rb2, rb3;
  ra  = *(const v8h*)(gA);
  rb0 = *(const v8h*)(gB + 0);
  rb1 = *(const v8h*)(gB + 8);
  rb2 = *(const v8h*)(gB + 16);
  rb3 = *(const v8h*)(gB + 24);
  *(v8h*)&As[0][a_row * 32 + a_col] = ra;
  *(v8h*)&Bs[0][tid * 32 +  0] = rb0;
  *(v8h*)&Bs[0][tid * 32 +  8] = rb1;
  *(v8h*)&Bs[0][tid * 32 + 16] = rb2;
  *(v8h*)&Bs[0][tid * 32 + 24] = rb3;
  __syncthreads();

  v8f a00 = {}, a01 = {}, a02 = {}, a03 = {};
  v8f a10 = {}, a11 = {}, a12 = {}, a13 = {};

  const int kn = K / 32;
  for (int ks = 0; ks < kn; ++ks) {
    const int cur = ks & 1;
    const bool more = (ks + 1 < kn);
    if (more) {
      const int k0 = (ks + 1) * 32;
      ra  = *(const v8h*)(gA + k0);
      rb0 = *(const v8h*)(gB + k0 + 0);
      rb1 = *(const v8h*)(gB + k0 + 8);
      rb2 = *(const v8h*)(gB + k0 + 16);
      rb3 = *(const v8h*)(gB + k0 + 24);
    }
    // issue all 12 ds_load_b128 into distinct registers, then run the WMMA chain
    const v16h af0 = load_frag_a (As[cur], 32, wr +  0, 0, lane);
    const v16h af1 = load_frag_a (As[cur], 32, wr + 16, 0, lane);
    const v16h bf0 = load_frag_bt(Bs[cur], 32, wc +  0, 0, lane);
    const v16h bf1 = load_frag_bt(Bs[cur], 32, wc + 16, 0, lane);
    const v16h bf2 = load_frag_bt(Bs[cur], 32, wc + 32, 0, lane);
    const v16h bf3 = load_frag_bt(Bs[cur], 32, wc + 48, 0, lane);
    a00 = WMMA_F16(a00, af0, bf0);
    a10 = WMMA_F16(a10, af1, bf0);
    a01 = WMMA_F16(a01, af0, bf1);
    a11 = WMMA_F16(a11, af1, bf1);
    a02 = WMMA_F16(a02, af0, bf2);
    a12 = WMMA_F16(a12, af1, bf2);
    a03 = WMMA_F16(a03, af0, bf3);
    a13 = WMMA_F16(a13, af1, bf3);
    if (more) {
      const int nxt = cur ^ 1;
      *(v8h*)&As[nxt][a_row * 32 + a_col] = ra;
      *(v8h*)&Bs[nxt][tid * 32 +  0] = rb0;
      *(v8h*)&Bs[nxt][tid * 32 +  8] = rb1;
      *(v8h*)&Bs[nxt][tid * 32 + 16] = rb2;
      *(v8h*)&Bs[nxt][tid * 32 + 24] = rb3;
      __syncthreads();
    }
  }
  store_tile_f16(C, ldc, rowB + wr +  0, colB + wc +  0, a00, lane);
  store_tile_f16(C, ldc, rowB + wr +  0, colB + wc + 16, a01, lane);
  store_tile_f16(C, ldc, rowB + wr +  0, colB + wc + 32, a02, lane);
  store_tile_f16(C, ldc, rowB + wr +  0, colB + wc + 48, a03, lane);
  store_tile_f16(C, ldc, rowB + wr + 16, colB + wc +  0, a10, lane);
  store_tile_f16(C, ldc, rowB + wr + 16, colB + wc + 16, a11, lane);
  store_tile_f16(C, ldc, rowB + wr + 16, colB + wc + 32, a12, lane);
  store_tile_f16(C, ldc, rowB + wr + 16, colB + wc + 48, a13, lane);
}

// ------- fused final GEMM: tanh([mix|Q] * W_out^T) + column sums -----------
// Same tiling; K runs over 2 phases (mix half then Q half of W_out columns).
__global__ __launch_bounds__(256) void gemm_out_tanh_colsum(const f16* __restrict__ Mx,
                                                            const f16* __restrict__ Qm,
                                                            const f16* __restrict__ W,
                                                            float* __restrict__ colsum) {
  __shared__ __align__(16) f16 As[2][64 * 32];
  __shared__ __align__(16) f16 Bs[2][256 * 32];
  const int tid  = threadIdx.x;
  const int lane = tid & 31;
  const int w    = tid >> 5;
  const int rowB = blockIdx.y * 64;   // d
  const int colB = blockIdx.x * 256;  // l
  const int wr   = (w & 1) * 32;
  const int wc   = (w >> 1) * 64;
  const int ldw  = 2 * Lh;
  const int kn   = Lh / 32;           // steps per phase

  const int a_row = tid >> 2, a_col = (tid & 3) * 8;
  const f16* gW = W + (size_t)(colB + tid) * ldw;

  auto loadA = [&](int ks) -> v8h {
    const int ph = (ks >= kn) ? 1 : 0;
    const int kk = (ks - ph * kn) * 32;
    const f16* src = ph ? Qm : Mx;
    return *(const v8h*)(src + (size_t)(rowB + a_row) * Lh + kk + a_col);
  };
  auto loadB = [&](int ks, int j) -> v8h {
    const int ph = (ks >= kn) ? 1 : 0;
    const int kk = ph * Lh + (ks - ph * kn) * 32;
    return *(const v8h*)(gW + kk + j * 8);
  };

  v8h ra, rb0, rb1, rb2, rb3;
  ra  = loadA(0);
  rb0 = loadB(0, 0); rb1 = loadB(0, 1); rb2 = loadB(0, 2); rb3 = loadB(0, 3);
  *(v8h*)&As[0][a_row * 32 + a_col] = ra;
  *(v8h*)&Bs[0][tid * 32 +  0] = rb0;
  *(v8h*)&Bs[0][tid * 32 +  8] = rb1;
  *(v8h*)&Bs[0][tid * 32 + 16] = rb2;
  *(v8h*)&Bs[0][tid * 32 + 24] = rb3;
  __syncthreads();

  v8f a00 = {}, a01 = {}, a02 = {}, a03 = {};
  v8f a10 = {}, a11 = {}, a12 = {}, a13 = {};

  const int total = 2 * kn;
  for (int ks = 0; ks < total; ++ks) {
    const int cur = ks & 1;
    const bool more = (ks + 1 < total);
    if (more) {
      ra  = loadA(ks + 1);
      rb0 = loadB(ks + 1, 0); rb1 = loadB(ks + 1, 1);
      rb2 = loadB(ks + 1, 2); rb3 = loadB(ks + 1, 3);
    }
    const v16h af0 = load_frag_a (As[cur], 32, wr +  0, 0, lane);
    const v16h af1 = load_frag_a (As[cur], 32, wr + 16, 0, lane);
    const v16h bf0 = load_frag_bt(Bs[cur], 32, wc +  0, 0, lane);
    const v16h bf1 = load_frag_bt(Bs[cur], 32, wc + 16, 0, lane);
    const v16h bf2 = load_frag_bt(Bs[cur], 32, wc + 32, 0, lane);
    const v16h bf3 = load_frag_bt(Bs[cur], 32, wc + 48, 0, lane);
    a00 = WMMA_F16(a00, af0, bf0);
    a10 = WMMA_F16(a10, af1, bf0);
    a01 = WMMA_F16(a01, af0, bf1);
    a11 = WMMA_F16(a11, af1, bf1);
    a02 = WMMA_F16(a02, af0, bf2);
    a12 = WMMA_F16(a12, af1, bf2);
    a03 = WMMA_F16(a03, af0, bf3);
    a13 = WMMA_F16(a13, af1, bf3);
    if (more) {
      const int nxt = cur ^ 1;
      *(v8h*)&As[nxt][a_row * 32 + a_col] = ra;
      *(v8h*)&Bs[nxt][tid * 32 +  0] = rb0;
      *(v8h*)&Bs[nxt][tid * 32 +  8] = rb1;
      *(v8h*)&Bs[nxt][tid * 32 + 16] = rb2;
      *(v8h*)&Bs[nxt][tid * 32 + 24] = rb3;
      __syncthreads();
    }
  }

  const int n = lane & 15;
  v8f* accs[8] = {&a00, &a01, &a02, &a03, &a10, &a11, &a12, &a13};
#pragma unroll
  for (int r = 0; r < 2; ++r)
#pragma unroll
    for (int c = 0; c < 4; ++c) {
      const v8f acc = *accs[r * 4 + c];
      float p = 0.f;
#pragma unroll
      for (int e = 0; e < 8; ++e) p += tanhf(acc[e]);
      atomicAdd(&colsum[colB + wc + 16 * c + n], p);
    }
}

// ---------------- conversions ----------------------------------------------
__global__ __launch_bounds__(256) void convert_tokens(const float* __restrict__ tok,
                                                      f16* __restrict__ tokH,
                                                      f16* __restrict__ ctxH) {
  __shared__ float tile[32][33];
  const int d0 = blockIdx.x * 32;
  const int l0 = blockIdx.y * 32;
  const int tx = threadIdx.x, ty = threadIdx.y;
#pragma unroll
  for (int i = 0; i < 32; i += 8) {
    float v = tok[(size_t)(l0 + ty + i) * Dc + (d0 + tx)];
    tile[ty + i][tx] = v;
    tokH[(size_t)(l0 + ty + i) * Dc + (d0 + tx)] = (f16)v;
  }
  __syncthreads();
#pragma unroll
  for (int i = 0; i < 32; i += 8)
    ctxH[(size_t)(d0 + ty + i) * Lh + (l0 + tx)] = (f16)tile[tx][ty + i];
}

__global__ void convert_f32_to_f16_x4(const float* __restrict__ src,
                                      f16* __restrict__ dst, int n4) {
  int i = blockIdx.x * blockDim.x + threadIdx.x;
  if (i < n4) {
    float4 v = ((const float4*)src)[i];
    v4h h = {(f16)v.x, (f16)v.y, (f16)v.z, (f16)v.w};
    *(v4h*)(dst + 4 * (size_t)i) = h;
  }
}

__global__ void zero_f32(float* p, int n) {
  int i = blockIdx.x * blockDim.x + threadIdx.x;
  if (i < n) p[i] = 0.f;
}

// ---------------- reductions ------------------------------------------------
__device__ __forceinline__ float block_reduce_max(float v, float* sm) {
  const int t = threadIdx.x;
  sm[t] = v; __syncthreads();
  for (int s = 128; s > 0; s >>= 1) {
    if (t < s) sm[t] = fmaxf(sm[t], sm[t + s]);
    __syncthreads();
  }
  float r = sm[0]; __syncthreads();
  return r;
}
__device__ __forceinline__ float block_reduce_sum(float v, float* sm) {
  const int t = threadIdx.x;
  sm[t] = v; __syncthreads();
  for (int s = 128; s > 0; s >>= 1) {
    if (t < s) sm[t] += sm[t + s];
    __syncthreads();
  }
  float r = sm[0]; __syncthreads();
  return r;
}

// In-place f16 row softmax over 4096 columns, one block per row.
__global__ __launch_bounds__(256) void softmax_rows_inplace(f16* __restrict__ S) {
  __shared__ float sm[256];
  const int t = threadIdx.x;
  f16* row = S + (size_t)blockIdx.x * Dc;
  float v[16];
#pragma unroll
  for (int j = 0; j < 16; ++j) v[j] = (float)row[t + 256 * j];
  float m = -3.4e38f;
#pragma unroll
  for (int j = 0; j < 16; ++j) m = fmaxf(m, v[j]);
  m = block_reduce_max(m, sm);
  float s = 0.f;
#pragma unroll
  for (int j = 0; j < 16; ++j) { v[j] = __expf(v[j] - m); s += v[j]; }
  s = block_reduce_sum(s, sm);
  const float inv = 1.0f / s;
#pragma unroll
  for (int j = 0; j < 16; ++j) row[t + 256 * j] = (f16)(v[j] * inv);
}

// scores -> /max -> softmax -> out[2048]. Single block.
__global__ __launch_bounds__(256) void finalize_scores(const float* __restrict__ colsum,
                                                       float* __restrict__ out) {
  __shared__ float sm[256];
  const int t = threadIdx.x;
  float v[8];
#pragma unroll
  for (int j = 0; j < 8; ++j) v[j] = colsum[t + 256 * j];
  float m = -3.4e38f;
#pragma unroll
  for (int j = 0; j < 8; ++j) m = fmaxf(m, v[j]);
  m = block_reduce_max(m, sm);                       // scores.max()
  float y[8];
  float m2l = -3.4e38f;
#pragma unroll
  for (int j = 0; j < 8; ++j) { y[j] = v[j] / m; m2l = fmaxf(m2l, y[j]); }
  const float m2 = block_reduce_max(m2l, sm);        // handles negative max
  float s = 0.f;
#pragma unroll
  for (int j = 0; j < 8; ++j) { y[j] = __expf(y[j] - m2); s += y[j]; }
  s = block_reduce_sum(s, sm);
  const float inv = 1.0f / s;
#pragma unroll
  for (int j = 0; j < 8; ++j) out[t + 256 * j] = y[j] * inv;
}

// ---------------------------------------------------------------------------
extern "C" void kernel_launch(void* const* d_in, const int* in_sizes, int n_in,
                              void* d_out, int out_size, void* d_ws, size_t ws_size,
                              hipStream_t stream) {
  (void)in_sizes; (void)n_in; (void)out_size; (void)ws_size;
  const float* tok  = (const float*)d_in[0];   // [2048, 4096]
  const float* Win  = (const float*)d_in[1];   // [2048, 2048]
  const float* Wout = (const float*)d_in[2];   // [2048, 4096]
  float* out = (float*)d_out;                  // [2048]

  char* ws = (char*)d_ws;
  size_t off = 0;
  auto take = [&](size_t bytes) -> char* {
    char* p = ws + off;
    off = (off + bytes + 255) & ~(size_t)255;
    return p;
  };
  f16*   ctxH   = (f16*)take((size_t)Dc * Lh * 2);      // context  [4096,2048]
  f16*   tokH   = (f16*)take((size_t)Lh * Dc * 2);      // tokens   [2048,4096]
  f16*   WinH   = (f16*)take((size_t)Lh * Lh * 2);      // W_in     [2048,2048]
  f16*   WoutH  = (f16*)take((size_t)Lh * 2 * Lh * 2);  // W_out    [2048,4096]
  f16*   Qh     = (f16*)take((size_t)Dc * Lh * 2);      // Q        [4096,2048]
  f16*   Sh     = (f16*)take((size_t)Dc * Dc * 2);      // S / A    [4096,4096]
  f16*   MixH   = (f16*)take((size_t)Dc * Lh * 2);      // mix      [4096,2048]
  float* colsum = (float*)take((size_t)Lh * 4);         // [2048]

  // 1) f32 -> f16 conversions (+ transpose for context)
  convert_tokens<<<dim3(Dc / 32, Lh / 32), dim3(32, 8), 0, stream>>>(tok, tokH, ctxH);
  convert_f32_to_f16_x4<<<(Lh * Lh / 4 + 255) / 256, 256, 0, stream>>>(Win, WinH, Lh * Lh / 4);
  convert_f32_to_f16_x4<<<(Lh * 2 * Lh / 4 + 255) / 256, 256, 0, stream>>>(Wout, WoutH, Lh * 2 * Lh / 4);
  zero_f32<<<Lh / 256, 256, 0, stream>>>(colsum, Lh);

  // 2) Q = context @ W_in^T       [4096,2048]
  gemm_nt_f16_lds<<<dim3(Lh / 256, Dc / 64), 256, 0, stream>>>(ctxH, Lh, WinH, Lh, Qh, Lh, Lh);
  // 3) S = Q @ context^T          [4096,4096]
  gemm_nt_f16_lds<<<dim3(Dc / 256, Dc / 64), 256, 0, stream>>>(Qh, Lh, ctxH, Lh, Sh, Dc, Lh);
  // 4) A = softmax_rows(S), in place
  softmax_rows_inplace<<<Dc, 256, 0, stream>>>(Sh);
  // 5) mix = A @ tokens^T (tokens == context^T)  [4096,2048]
  gemm_nt_f16_lds<<<dim3(Lh / 256, Dc / 64), 256, 0, stream>>>(Sh, Dc, tokH, Dc, MixH, Lh, Dc);
  // 6) out = tanh([mix|Q] @ W_out^T), column sums into colsum
  gemm_out_tanh_colsum<<<dim3(Lh / 256, Dc / 64), 256, 0, stream>>>(MixH, Qh, WoutH, colsum);
  // 7) scores /= max; softmax -> d_out
  finalize_scores<<<1, 256, 0, stream>>>(colsum, out);
}
